// MultiHeadAttention_77257871720722
// MI455X (gfx1250) — matrix operand order repaired and checked by reading, not compiled, not run on hardware
//
#include <hip/hip_runtime.h>
#include <hip/hip_bf16.h>

// ---------------------------------------------------------------------------
// MHA forward for gfx1250 (MI455X): bf16 WMMA everywhere, f32 accum/softmax.
// B=4, S=2048, D=1024, H=16, HD=64
// ---------------------------------------------------------------------------

typedef __attribute__((ext_vector_type(16))) __bf16 v16bf;
typedef __attribute__((ext_vector_type(8)))  __bf16 v8bf;
typedef __attribute__((ext_vector_type(8)))  float  v8f;

constexpr int Bc  = 4;
constexpr int Sc  = 2048;
constexpr int Dc  = 1024;
constexpr int Hc  = 16;
constexpr int HDc = 64;

#define V8F_ZERO ((v8f){0,0,0,0,0,0,0,0})

__device__ inline v8f wmma_bf16(v16bf a, v16bf b, v8f c) {
  // D = A(16x32 bf16) * B(32x16 bf16) + C(16x16 f32)
  return __builtin_amdgcn_wmma_f32_16x16x32_bf16(
      /*neg_a=*/false, a, /*neg_b=*/false, b,
      /*c_mod=*/(short)0, c, /*reuse_a=*/false, /*reuse_b=*/false);
}

// 16 consecutive f32 -> bf16 B-fragment slice (lane already offset).
__device__ inline v16bf cvt16_f32(const float* __restrict__ p) {
  float4 f0 = *(const float4*)(p + 0);
  float4 f1 = *(const float4*)(p + 4);
  float4 f2 = *(const float4*)(p + 8);
  float4 f3 = *(const float4*)(p + 12);
  v16bf r;
  r[0]=(__bf16)f0.x; r[1]=(__bf16)f0.y; r[2]=(__bf16)f0.z; r[3]=(__bf16)f0.w;
  r[4]=(__bf16)f1.x; r[5]=(__bf16)f1.y; r[6]=(__bf16)f1.z; r[7]=(__bf16)f1.w;
  r[8]=(__bf16)f2.x; r[9]=(__bf16)f2.y; r[10]=(__bf16)f2.z; r[11]=(__bf16)f2.w;
  r[12]=(__bf16)f3.x; r[13]=(__bf16)f3.y; r[14]=(__bf16)f3.z; r[15]=(__bf16)f3.w;
  return r;
}

// A-fragment (16x32, 16-bit): lane half=lane>>4, m=lane&15.
// elems j: K = half*16+j (contiguous). Two 8-float runs.
__device__ inline v16bf load_a_f32(const float* __restrict__ row, int half) {
  const float* p0 = row + half * 16;
  const float* p1 = row + half * 16 + 8;
  float4 a0 = *(const float4*)(p0 + 0);
  float4 a1 = *(const float4*)(p0 + 4);
  float4 b0 = *(const float4*)(p1 + 0);
  float4 b1 = *(const float4*)(p1 + 4);
  v16bf r;
  r[0]=(__bf16)a0.x; r[1]=(__bf16)a0.y; r[2]=(__bf16)a0.z; r[3]=(__bf16)a0.w;
  r[4]=(__bf16)a1.x; r[5]=(__bf16)a1.y; r[6]=(__bf16)a1.z; r[7]=(__bf16)a1.w;
  r[8]=(__bf16)b0.x; r[9]=(__bf16)b0.y; r[10]=(__bf16)b0.z; r[11]=(__bf16)b0.w;
  r[12]=(__bf16)b1.x; r[13]=(__bf16)b1.y; r[14]=(__bf16)b1.z; r[15]=(__bf16)b1.w;
  return r;
}

// Same A layout but from a bf16 row-major source (e.g. Q, or P in LDS).
__device__ inline v16bf load_a_bf16(const __bf16* __restrict__ row, int half) {
  v8bf lo = *(const v8bf*)(row + half * 16);
  v8bf hi = *(const v8bf*)(row + half * 16 + 8);
  v16bf r;
#pragma unroll
  for (int j = 0; j < 8; ++j) { r[j] = lo[j]; r[8 + j] = hi[j]; }
  return r;
}

// Row-max reduction within each 16-lane half (C-tile rows live per-half).
__device__ inline float rmax16(float v) {
  v = fmaxf(v, __shfl_xor(v, 1, 32));
  v = fmaxf(v, __shfl_xor(v, 2, 32));
  v = fmaxf(v, __shfl_xor(v, 4, 32));
  v = fmaxf(v, __shfl_xor(v, 8, 32));
  return v;
}

// ---------------------------------------------------------------------------
// GEMM: C[M=8192, N=1024] = scale * (A[M,1024](f32) @ W[N,1024]^T), bf16 WMMA.
// MODE 0: out bf16 [B,H,S,HD]   (Q with folded softmax scale, K)
// MODE 1: out bf16 [B,H,HD,S]   (V transposed)
// MODE 2: out f32  [M,N]        (final projection)
// Block: 256 thr = 8 waves in 4(m) x 2(n); wave tile 32x64; block tile 128x128.
// ---------------------------------------------------------------------------
template <int MODE>
__global__ __launch_bounds__(256) void gemm_xwT_kernel(
    const float* __restrict__ A, const float* __restrict__ W,
    void* __restrict__ Out, float out_scale) {
  const int K = Dc;
  const int tid  = threadIdx.x;
  const int lane = tid & 31;
  const int wid  = tid >> 5;
  const int half = lane >> 4;
  const int nn   = lane & 15;
  const int m_base = blockIdx.y * 128 + (wid >> 1) * 32;
  const int n_base = blockIdx.x * 128 + (wid & 1) * 64;

  v8f acc[2][4];
#pragma unroll
  for (int i = 0; i < 2; ++i)
#pragma unroll
    for (int j = 0; j < 4; ++j) acc[i][j] = V8F_ZERO;

  const float* arow0 = A + (size_t)(m_base + nn) * K;
  const float* arow1 = A + (size_t)(m_base + 16 + nn) * K;

  for (int k0 = 0; k0 < K; k0 += 32) {
    __builtin_prefetch(arow0 + k0 + 32, 0, 0);  // global_prefetch_b8
    __builtin_prefetch(arow1 + k0 + 32, 0, 0);
    v16bf a0 = load_a_f32(arow0 + k0, half);
    v16bf a1 = load_a_f32(arow1 + k0, half);
#pragma unroll
    for (int nt = 0; nt < 4; ++nt) {
      const float* wp = W + (size_t)(n_base + nt * 16 + nn) * K + k0 + half * 16;
      v16bf bw = cvt16_f32(wp);
      acc[0][nt] = wmma_bf16(a0, bw, acc[0][nt]);
      acc[1][nt] = wmma_bf16(a1, bw, acc[1][nt]);
    }
  }

#pragma unroll
  for (int mt = 0; mt < 2; ++mt) {
#pragma unroll
    for (int nt = 0; nt < 4; ++nt) {
#pragma unroll
      for (int r = 0; r < 8; ++r) {
        const int m = m_base + mt * 16 + half * 8 + r;
        const int n = n_base + nt * 16 + nn;
        const float v = acc[mt][nt][r] * out_scale;
        if constexpr (MODE == 2) {
          ((float*)Out)[(size_t)m * Dc + n] = v;
        } else {
          const int b = m >> 11, s = m & (Sc - 1);   // S = 2048
          const int h = n >> 6,  hd = n & (HDc - 1); // HD = 64
          if constexpr (MODE == 0)
            ((__bf16*)Out)[((size_t)(b * Hc + h) * Sc + s) * HDc + hd] = (__bf16)v;
          else
            ((__bf16*)Out)[((size_t)(b * Hc + h) * HDc + hd) * Sc + s] = (__bf16)v;
        }
      }
    }
  }
}

// ---------------------------------------------------------------------------
// Flash attention core: one 32-key block against a 32-row q tile.
// MASK=true only for the diagonal block; all others are branch/cndmask-free.
// Scores arrive pre-scaled into the log2 domain (scale folded into Q), so the
// softmax uses raw exp2. Row-sum of P is computed with a WMMA against ones.
// ---------------------------------------------------------------------------
template <bool MASK>
__device__ inline void attn_block(
    int kb, int q0, int half, int nn,
    const __bf16* __restrict__ Kh, const __bf16* __restrict__ Vh,
    const v16bf (&aQ)[2][2], v16bf ones,
    float (&mrow)[2][8], v8f (&lacc)[2], v8f (&o)[2][4],
    __bf16 (*__restrict__ pw)[32]) {
  // K fragments (shared by both m tiles): 4 x 32B contiguous loads.
  const __bf16* kp0 = Kh + (size_t)(kb + nn) * HDc + half * 16;
  const __bf16* kp1 = Kh + (size_t)(kb + 16 + nn) * HDc + half * 16;
  v16bf bK00 = *(const v16bf*)(kp0);        // hd 0..31
  v16bf bK01 = *(const v16bf*)(kp0 + 32);   // hd 32..63
  v16bf bK10 = *(const v16bf*)(kp1);
  v16bf bK11 = *(const v16bf*)(kp1 + 32);

#pragma unroll
  for (int mt = 0; mt < 2; ++mt) {
    v8f s0 = V8F_ZERO, s1 = V8F_ZERO;
    s0 = wmma_bf16(aQ[mt][0], bK00, s0);
    s0 = wmma_bf16(aQ[mt][1], bK01, s0);
    s1 = wmma_bf16(aQ[mt][0], bK10, s1);
    s1 = wmma_bf16(aQ[mt][1], bK11, s1);

    float alpha[8];
#pragma unroll
    for (int r = 0; r < 8; ++r) {
      float v0 = s0[r];
      float v1 = s1[r];
      if constexpr (MASK) {
        const int qi = q0 + mt * 16 + half * 8 + r;
        if (kb + nn > qi)      v0 = -1e30f;   // exp2(-1e30) == 0
        if (kb + 16 + nn > qi) v1 = -1e30f;
      }
      const float rm = rmax16(fmaxf(v0, v1));
      const float mn = fmaxf(mrow[mt][r], rm);
      const float al = exp2f(mrow[mt][r] - mn);
      const float p0 = exp2f(v0 - mn);
      const float p1 = exp2f(v1 - mn);
      mrow[mt][r] = mn;
      alpha[r] = al;
      pw[mt * 16 + half * 8 + r][nn]      = (__bf16)p0;
      pw[mt * 16 + half * 8 + r][16 + nn] = (__bf16)p1;
    }
#pragma unroll
    for (int r = 0; r < 8; ++r) lacc[mt][r] *= alpha[r];
#pragma unroll
    for (int j4 = 0; j4 < 4; ++j4)
#pragma unroll
      for (int r = 0; r < 8; ++r) o[mt][j4][r] *= alpha[r];
  }

  asm volatile("s_wait_dscnt 0" ::: "memory");  // P stores -> P loads (same wave)

  // P A-fragments (32x32 tile -> two 16x32 fragments) from LDS.
  v16bf aP0 = load_a_bf16(&pw[nn][0], half);
  v16bf aP1 = load_a_bf16(&pw[16 + nn][0], half);

  // l += rowsum(P) via WMMA against an all-ones B (replicated C layout).
  lacc[0] = wmma_bf16(aP0, ones, lacc[0]);
  lacc[1] = wmma_bf16(aP1, ones, lacc[1]);

  // O += P (32x32) @ V (32x64): V fragments shared by both m tiles.
#pragma unroll
  for (int j4 = 0; j4 < 4; ++j4) {
    v16bf bV = *(const v16bf*)(Vh + (size_t)(j4 * 16 + nn) * Sc + kb + half * 16);
    o[0][j4] = wmma_bf16(aP0, bV, o[0][j4]);
    o[1][j4] = wmma_bf16(aP1, bV, o[1][j4]);
  }
}

// ---------------------------------------------------------------------------
// Flash attention (causal, online softmax). One wave owns 32 query rows.
// Q,K: bf16 [B,H,S,HD] (Q pre-scaled by log2e/sqrt(HD)); Vt: bf16 [B,H,HD,S];
// Out: f32 [B,S,D].
// ---------------------------------------------------------------------------
__global__ __launch_bounds__(256) void flash_attn_kernel(
    const __bf16* __restrict__ Q, const __bf16* __restrict__ Kt,
    const __bf16* __restrict__ Vt, float* __restrict__ Outp) {
  __shared__ __bf16 plds[8][32][32];  // per-wave P scratch (2 KB each)

  const int tid  = threadIdx.x;
  const int lane = tid & 31;
  const int w    = tid >> 5;
  const int half = lane >> 4;
  const int nn   = lane & 15;

  const int qt = blockIdx.x * 8 + w;          // 4096 q-tiles of 32 rows
  const int q0 = (qt & (Sc / 32 - 1)) * 32;   // query row base
  const int bh = qt >> 6;                     // (b*H + h)

  const __bf16* Qh = Q  + (size_t)bh * Sc * HDc;
  const __bf16* Kh = Kt + (size_t)bh * Sc * HDc;
  const __bf16* Vh = Vt + (size_t)bh * HDc * Sc;
  __bf16 (*pw)[32] = plds[w];

  // Q A-fragments: 2 m-tiles x 2 hd k-steps, loaded once.
  v16bf aQ[2][2];
#pragma unroll
  for (int mt = 0; mt < 2; ++mt) {
    const __bf16* qrow = Qh + (size_t)(q0 + mt * 16 + nn) * HDc;
    aQ[mt][0] = load_a_bf16(qrow, half);
    aQ[mt][1] = load_a_bf16(qrow + 32, half);
  }

  v16bf ones;
#pragma unroll
  for (int j = 0; j < 16; ++j) ones[j] = (__bf16)1.0f;

  float mrow[2][8];
  v8f lacc[2];
  v8f o[2][4];
#pragma unroll
  for (int mt = 0; mt < 2; ++mt) {
    lacc[mt] = V8F_ZERO;
#pragma unroll
    for (int r = 0; r < 8; ++r) mrow[mt][r] = -1e30f;
#pragma unroll
    for (int j = 0; j < 4; ++j) o[mt][j] = V8F_ZERO;
  }

  // Full (unmasked) blocks strictly below the diagonal, then diagonal block.
  for (int kb = 0; kb < q0; kb += 32)
    attn_block<false>(kb, q0, half, nn, Kh, Vh, aQ, ones, mrow, lacc, o, pw);
  attn_block<true>(q0, q0, half, nn, Kh, Vh, aQ, ones, mrow, lacc, o, pw);

  // Epilogue: O /= l, write f32 [B,S,D].
  const int b = bh >> 4, h = bh & (Hc - 1);
#pragma unroll
  for (int mt = 0; mt < 2; ++mt) {
#pragma unroll
    for (int r = 0; r < 8; ++r) {
      const float inv = 1.0f / lacc[mt][r];
      const int srow = q0 + mt * 16 + half * 8 + r;
#pragma unroll
      for (int j4 = 0; j4 < 4; ++j4) {
        Outp[((size_t)(b * Sc + srow)) * Dc + h * HDc + j4 * 16 + nn] =
            o[mt][j4][r] * inv;
      }
    }
  }
}

// ---------------------------------------------------------------------------
// Launcher. Inputs: x, wq, wk, wv, wo (all f32). Output: f32 [B,S,D].
// Workspace: Q bf16 | K bf16 | Vt bf16 | attn f32  (~80 MB).
// ---------------------------------------------------------------------------
extern "C" void kernel_launch(void* const* d_in, const int* in_sizes, int n_in,
                              void* d_out, int out_size, void* d_ws, size_t ws_size,
                              hipStream_t stream) {
  const float* x  = (const float*)d_in[0];
  const float* wq = (const float*)d_in[1];
  const float* wk = (const float*)d_in[2];
  const float* wv = (const float*)d_in[3];
  const float* wo = (const float*)d_in[4];

  const size_t elems = (size_t)Bc * Sc * Dc;       // 8,388,608
  char* ws = (char*)d_ws;
  __bf16* qb   = (__bf16*)(ws);
  __bf16* kb   = (__bf16*)(ws + elems * 2);
  __bf16* vtb  = (__bf16*)(ws + elems * 4);
  float*  attn = (float*)(ws + elems * 6);

  dim3 gblk(256);
  dim3 ggrid(Dc / 128, (Bc * Sc) / 128);           // (8, 64)

  // Q gets softmax scale folded in, converted to the log2 domain:
  // scores_log2 = (q . k) * (1/sqrt(HD)) * log2(e)
  const float q_scale = 0.125f * 1.4426950408889634f;

  gemm_xwT_kernel<0><<<ggrid, gblk, 0, stream>>>(x, wq, qb, q_scale);
  gemm_xwT_kernel<0><<<ggrid, gblk, 0, stream>>>(x, wk, kb, 1.0f);
  gemm_xwT_kernel<1><<<ggrid, gblk, 0, stream>>>(x, wv, vtb, 1.0f);

  dim3 agrid((Bc * Hc * (Sc / 32)) / 8);           // 512 blocks x 8 waves
  flash_attn_kernel<<<agrid, gblk, 0, stream>>>(qb, kb, vtb, attn);

  gemm_xwT_kernel<2><<<ggrid, gblk, 0, stream>>>(attn, wo, d_out, 1.0f);
}
